// Qwen2_5_VLModel_24086176596556
// MI455X (gfx1250) — matrix operand order repaired and checked by reading, not compiled, not run on hardware
//
#include <hip/hip_runtime.h>

// ---------------- constants (from reference) ----------------
#define TTOT   4096      // total tokens
#define HIDDEN 3584
#define NH     28
#define NKV    4
#define HD     128
#define SEQ    1024
#define NB     4         // batches (TTOT / SEQ)

typedef _Float16 f16;
typedef __attribute__((ext_vector_type(8)))  _Float16 v8h;
typedef __attribute__((ext_vector_type(16))) _Float16 v16h;
typedef __attribute__((ext_vector_type(8)))  float    v8f;

// ---------------- WMMA fragment loaders (ISA 7.12.2 layouts) ----------------
// A (16x32 f16, M x K): lane holds row m = lane&15.
//   lanes 0-15 : elems 0..7 -> K = k0+0..7,  elems 8..15 -> K = k0+16..23
//   lanes 16-31: elems 0..7 -> K = k0+8..15, elems 8..15 -> K = k0+24..31
__device__ __forceinline__ v16h load_a32(const f16* __restrict__ base, int stride,
                                         int k0, int lane) {
  const f16* p = base + (size_t)(lane & 15) * stride + k0 + ((lane >> 4) << 3);
  v8h lo = *(const v8h*)p;          // 16B
  v8h hi = *(const v8h*)(p + 16);   // 16B
  v16h r;
#pragma unroll
  for (int i = 0; i < 8; ++i) { r[i] = lo[i]; r[i + 8] = hi[i]; }
  return r;
}

// B (32x16 f16, K x N) taken from Bt stored row-major [N, K]:
//   lane holds column n = n0 + (lane&15); same K order as A:
//   elems 0..7 -> K = k0+(lane>>4)*8+0..7, elems 8..15 -> K = k0+16+(lane>>4)*8+0..7
__device__ __forceinline__ v16h load_b32(const f16* __restrict__ bt, int stride,
                                         int n0, int k0, int lane) {
  const f16* p = bt + (size_t)(n0 + (lane & 15)) * stride + k0 + ((lane >> 4) << 3);
  v8h lo = *(const v8h*)p;          // 16B
  v8h hi = *(const v8h*)(p + 16);   // 16B
  v16h r;
#pragma unroll
  for (int i = 0; i < 8; ++i) { r[i] = lo[i]; r[i + 8] = hi[i]; }
  return r;
}

// ---------------- prep kernels ----------------
__global__ void cvt_f32_to_f16(const float* __restrict__ in, f16* __restrict__ out, long n) {
  long i = (long)blockIdx.x * blockDim.x + threadIdx.x;
  if (i < n) out[i] = (f16)in[i];
}

// out[c*R + r] = (f16) in[r*C + c]   (in: [R,C] fp32 -> out: [C,R] f16)
__global__ void transpose_f32_to_f16(const float* __restrict__ in, f16* __restrict__ out,
                                     int R, int C) {
  long i = (long)blockIdx.x * blockDim.x + threadIdx.x;
  if (i >= (long)R * C) return;
  int c = (int)(i % C);
  int r = (int)(i / C);
  out[(size_t)c * R + r] = (f16)in[i];
}

__global__ void transpose_f16_kernel(const f16* __restrict__ in, f16* __restrict__ out,
                                     int R, int C) {
  long i = (long)blockIdx.x * blockDim.x + threadIdx.x;
  if (i >= (long)R * C) return;
  int c = (int)(i % C);
  int r = (int)(i / C);
  out[(size_t)c * R + r] = in[i];
}

// ---------------- WMMA GEMM: C[M,N] = A[M,K] * Bt[N,K]^T (+bias) ----------------
// block = 256 threads = 8 waves; wave w -> rows m0 = by*128 + w*16, cols n0 = bx*64
template <typename OutT>
__global__ __launch_bounds__(256) void gemm_wmma_kernel(
    const f16* __restrict__ A, const f16* __restrict__ Bt,
    const float* __restrict__ bias, OutT* __restrict__ C,
    int M, int N, int K) {
  const int lane = threadIdx.x & 31;
  const int wv   = threadIdx.x >> 5;
  const int m0   = blockIdx.y * 128 + wv * 16;
  const int n0   = blockIdx.x * 64;
  const f16* Abase = A + (size_t)m0 * K;

  v8f acc[4] = {};
  for (int k0 = 0; k0 < K; k0 += 32) {
    v16h a = load_a32(Abase, K, k0, lane);
#pragma unroll
    for (int j = 0; j < 4; ++j) {
      v16h b = load_b32(Bt, K, n0 + 16 * j, k0, lane);
      acc[j] = __builtin_amdgcn_wmma_f32_16x16x32_f16(
          false, a, false, b, (short)0, acc[j], false, false);
    }
  }
  const int n     = lane & 15;
  const int rbase = (lane >> 4) << 3;   // 0 or 8
#pragma unroll
  for (int j = 0; j < 4; ++j) {
    int nn = n0 + 16 * j + n;
    float bval = bias ? bias[nn] : 0.0f;
#pragma unroll
    for (int r = 0; r < 8; ++r) {
      C[(size_t)(m0 + rbase + r) * N + nn] = (OutT)(acc[j][r] + bval);
    }
  }
}

// ---------------- MRoPE ----------------
// sections [16,24,24]*2 over head dim 128; pattern repeats every 64 lanes.
__device__ __forceinline__ int mrope_sec(int d) {
  int dd = d & 63;
  return (dd < 16) ? 0 : ((dd < 40) ? 1 : 2);
}

__global__ void rope_kernel(f16* __restrict__ q, f16* __restrict__ kk,
                            const float* __restrict__ cosT, const float* __restrict__ sinT) {
  long idx = (long)blockIdx.x * blockDim.x + threadIdx.x;
  long total = (long)TTOT * (NH + NKV) * 64;
  if (idx >= total) return;
  int  d    = (int)(idx & 63);
  long rest = idx >> 6;
  int  head = (int)(rest % (NH + NKV));
  int  t    = (int)(rest / (NH + NKV));
  f16* ptr = (head < NH)
      ? (q  + ((size_t)t * NH  + head) * HD)
      : (kk + ((size_t)t * NKV + (head - NH)) * HD);
  int d2 = d + 64;                      // sec(d2) == sec(d)
  const float* ct = cosT + (size_t)mrope_sec(d) * TTOT * HD + (size_t)t * HD;
  const float* st = sinT + (size_t)mrope_sec(d) * TTOT * HD + (size_t)t * HD;
  float c1 = ct[d],  s1 = st[d];
  float c2 = ct[d2], s2 = st[d2];
  float x1 = (float)ptr[d], x2 = (float)ptr[d2];
  ptr[d]  = (f16)(x1 * c1 - x2 * s1);   // x1*cos + (-x2)*sin
  ptr[d2] = (f16)(x2 * c2 + x1 * s2);   // x2*cos + ( x1)*sin
}

// ---------------- flash attention (causal, GQA) ----------------
__device__ __forceinline__ float rmax16(float v) {
  v = fmaxf(v, __shfl_xor(v, 1, 32));
  v = fmaxf(v, __shfl_xor(v, 2, 32));
  v = fmaxf(v, __shfl_xor(v, 4, 32));
  v = fmaxf(v, __shfl_xor(v, 8, 32));
  return v;
}
__device__ __forceinline__ float rsum16(float v) {
  v += __shfl_xor(v, 1, 32);
  v += __shfl_xor(v, 2, 32);
  v += __shfl_xor(v, 4, 32);
  v += __shfl_xor(v, 8, 32);
  return v;
}

// one wave handles (batch b, head h, 16 query rows); key blocks of 32
__global__ __launch_bounds__(128) void attn_kernel(
    const f16* __restrict__ q,   // [TTOT, NH, HD]
    const f16* __restrict__ k,   // [TTOT, NKV, HD]
    const f16* __restrict__ vt,  // [NKV*HD, TTOT]  (V transposed)
    f16* __restrict__ out) {     // [TTOT, NH*HD]
  __shared__ _Float16 ptile[4][16][32];   // per-wave P staging (C-layout -> A-layout)
  const int lane = threadIdx.x & 31;
  const int wv   = threadIdx.x >> 5;
  int gw = blockIdx.x * 4 + wv;
  const int QT = SEQ / 16;
  int qt = gw % QT;  gw /= QT;
  int h  = gw % NH;
  int b  = gw / NH;
  int kvh = h / (NH / NKV);

  // preload Q fragments for this 16-row tile (full head dim, 4 k-steps)
  const f16* qbase = q + ((size_t)(b * SEQ + qt * 16) * NH + h) * HD;
  v16h qa[4];
#pragma unroll
  for (int i = 0; i < 4; ++i) qa[i] = load_a32(qbase, NH * HD, i * 32, lane);

  v8f   o[8] = {};
  float mi[8], li[8];
#pragma unroll
  for (int r = 0; r < 8; ++r) { mi[r] = -1e30f; li[r] = 0.0f; }

  const float scale = 0.088388347648318447f;   // 1/sqrt(128)
  const int row_base = qt * 16 + ((lane >> 4) << 3);  // + r = local query row
  const int kend = qt * 16 + 16;

  for (int kb = 0; kb < kend; kb += 32) {
    // ---- scores: two 16x16 tiles over keys [kb, kb+32) ----
    float sc[2][8];
#pragma unroll
    for (int half = 0; half < 2; ++half) {
      int kt = kb + half * 16;
      const f16* kbase = k + ((size_t)(b * SEQ + kt) * NKV + kvh) * HD;
      v8f s = {};
#pragma unroll
      for (int d0 = 0; d0 < 4; ++d0) {
        v16h bf = load_b32(kbase, NKV * HD, 0, d0 * 32, lane);
        s = __builtin_amdgcn_wmma_f32_16x16x32_f16(
            false, qa[d0], false, bf, (short)0, s, false, false);
      }
      int key = kt + (lane & 15);
#pragma unroll
      for (int r = 0; r < 8; ++r)
        sc[half][r] = (key > row_base + r) ? -1e30f : s[r] * scale;
    }

    // ---- online softmax (row stats reduced across 16-lane halves) ----
    float pr[2][8];
#pragma unroll
    for (int r = 0; r < 8; ++r) {
      float tmax = rmax16(fmaxf(sc[0][r], sc[1][r]));
      float mnew = fmaxf(mi[r], tmax);
      float alpha = __expf(mi[r] - mnew);
      float p0 = __expf(sc[0][r] - mnew);
      float p1 = __expf(sc[1][r] - mnew);
      li[r] = li[r] * alpha + rsum16(p0 + p1);
      mi[r] = mnew;
      pr[0][r] = p0;  pr[1][r] = p1;
#pragma unroll
      for (int dt = 0; dt < 8; ++dt) o[dt][r] *= alpha;
    }

    // ---- re-layout P (C-layout) -> A-layout via per-wave LDS slice ----
    {
      int n = lane & 15;
#pragma unroll
      for (int r = 0; r < 8; ++r) {
        int row = r + ((lane >> 4) << 3);
        ptile[wv][row][n]      = (_Float16)pr[0][r];
        ptile[wv][row][16 + n] = (_Float16)pr[1][r];
      }
    }
    asm volatile("s_wait_dscnt 0" ::: "memory");   // wave-local DS write->read ordering
    v16h pa;
    {
      const _Float16* p = &ptile[wv][lane & 15][(lane >> 4) << 3];
      v8h lo = *(const v8h*)p;
      v8h hi = *(const v8h*)(p + 16);
#pragma unroll
      for (int i = 0; i < 8; ++i) { pa[i] = lo[i]; pa[i + 8] = hi[i]; }
    }

    // ---- O += P(16x32) * V(32x128), V read from transposed layout ----
    const f16* vbase = vt + (size_t)(kvh * HD) * TTOT + (size_t)b * SEQ + kb;
#pragma unroll
    for (int dt = 0; dt < 8; ++dt) {
      v16h bf = load_b32(vbase + (size_t)(dt * 16) * TTOT, TTOT, 0, 0, lane);
      o[dt] = __builtin_amdgcn_wmma_f32_16x16x32_f16(
          false, pa, false, bf, (short)0, o[dt], false, false);
    }
  }

  // ---- epilogue: normalize and store ----
  int n = lane & 15;
#pragma unroll
  for (int r = 0; r < 8; ++r) {
    float inv = 1.0f / li[r];
    int t = b * SEQ + qt * 16 + r + ((lane >> 4) << 3);
    f16* orow = out + (size_t)t * (NH * HD) + h * HD;
#pragma unroll
    for (int dt = 0; dt < 8; ++dt)
      orow[dt * 16 + n] = (f16)(o[dt][r] * inv);
  }
}

// ---------------- host launch ----------------
extern "C" void kernel_launch(void* const* d_in, const int* in_sizes, int n_in,
                              void* d_out, int out_size, void* d_ws, size_t ws_size,
                              hipStream_t stream) {
  const float* hs   = (const float*)d_in[0];
  const float* cosT = (const float*)d_in[1];
  const float* sinT = (const float*)d_in[2];
  // d_in[3..5] = cu_seqlens / max_seqlen / position_ids (constants hardcoded)
  const float* Wq = (const float*)d_in[6];
  const float* bq = (const float*)d_in[7];
  const float* Wk = (const float*)d_in[8];
  const float* bk = (const float*)d_in[9];
  const float* Wv = (const float*)d_in[10];
  const float* bv = (const float*)d_in[11];
  const float* Wo = (const float*)d_in[12];
  float* outp = (float*)d_out;

  char* ws = (char*)d_ws;
  size_t off = 0;
  auto carve = [&](size_t bytes) -> char* {
    char* p = ws + off;
    off += (bytes + 255) & ~(size_t)255;
    return p;
  };
  f16* hid16 = (f16*)carve((size_t)TTOT * HIDDEN * 2);          // later reused as attn-out
  f16* wqT   = (f16*)carve((size_t)HIDDEN * (NH  * HD) * 2);    // later reused as Wo^T
  f16* wkT   = (f16*)carve((size_t)HIDDEN * (NKV * HD) * 2);
  f16* wvT   = (f16*)carve((size_t)HIDDEN * (NKV * HD) * 2);
  f16* qbuf  = (f16*)carve((size_t)TTOT * (NH  * HD) * 2);
  f16* kbuf  = (f16*)carve((size_t)TTOT * (NKV * HD) * 2);
  f16* vbuf  = (f16*)carve((size_t)TTOT * (NKV * HD) * 2);
  f16* vtbuf = (f16*)carve((size_t)TTOT * (NKV * HD) * 2);
  f16* attn  = hid16;   // alias: hidden_f16 dead after QKV GEMMs
  f16* woT   = wqT;     // alias: Wq^T dead after Q GEMM (same size)

  // 1) convert hidden states to f16
  { long n = (long)TTOT * HIDDEN;
    cvt_f32_to_f16<<<dim3((unsigned)((n + 255) / 256)), 256, 0, stream>>>(hs, hid16, n); }

  // 2) stage transposed f16 weights (B-fragments become contiguous)
  { long n = (long)HIDDEN * (NH * HD);
    transpose_f32_to_f16<<<dim3((unsigned)((n + 255) / 256)), 256, 0, stream>>>(Wq, wqT, HIDDEN, NH * HD); }
  { long n = (long)HIDDEN * (NKV * HD);
    transpose_f32_to_f16<<<dim3((unsigned)((n + 255) / 256)), 256, 0, stream>>>(Wk, wkT, HIDDEN, NKV * HD);
    transpose_f32_to_f16<<<dim3((unsigned)((n + 255) / 256)), 256, 0, stream>>>(Wv, wvT, HIDDEN, NKV * HD); }

  // 3) QKV projections (WMMA, bias fused)
  gemm_wmma_kernel<f16><<<dim3((NH * HD) / 64, TTOT / 128), 256, 0, stream>>>(
      hid16, wqT, bq, qbuf, TTOT, NH * HD, HIDDEN);
  gemm_wmma_kernel<f16><<<dim3((NKV * HD) / 64, TTOT / 128), 256, 0, stream>>>(
      hid16, wkT, bk, kbuf, TTOT, NKV * HD, HIDDEN);
  gemm_wmma_kernel<f16><<<dim3((NKV * HD) / 64, TTOT / 128), 256, 0, stream>>>(
      hid16, wvT, bv, vbuf, TTOT, NKV * HD, HIDDEN);

  // 4) Wo^T into the (now free) Wq^T slot
  { long n = (long)(NH * HD) * HIDDEN;
    transpose_f32_to_f16<<<dim3((unsigned)((n + 255) / 256)), 256, 0, stream>>>(Wo, woT, NH * HD, HIDDEN); }

  // 5) MRoPE on Q and K
  { long n = (long)TTOT * (NH + NKV) * 64;
    rope_kernel<<<dim3((unsigned)((n + 255) / 256)), 256, 0, stream>>>(qbuf, kbuf, cosT, sinT); }

  // 6) V -> V^T so P*V B-fragments are contiguous
  { long n = (long)TTOT * (NKV * HD);
    transpose_f16_kernel<<<dim3((unsigned)((n + 255) / 256)), 256, 0, stream>>>(vbuf, vtbuf, TTOT, NKV * HD); }

  // 7) causal GQA flash attention (1 wave per 16-query tile; 4 waves/block)
  attn_kernel<<<dim3(NB * NH * (SEQ / 16) / 4), 128, 0, stream>>>(qbuf, kbuf, vtbuf, attn);

  // 8) output projection -> fp32 result
  gemm_wmma_kernel<float><<<dim3(HIDDEN / 64, TTOT / 128), 256, 0, stream>>>(
      attn, woT, nullptr, outp, TTOT, HIDDEN, NH * HD);
}
